// MultiHeadAttention_29489245455122
// MI455X (gfx1250) — compile-verified, matrix operand order checked by reading, not run on hardware
//
#include <hip/hip_runtime.h>

// Problem constants (match reference)
#define BB 2
#define SS 2048
#define DD 1024
#define HH 16
#define DH 64

typedef __bf16 bf16_t;
typedef bf16_t v16bf __attribute__((ext_vector_type(16)));
typedef float  v8f   __attribute__((ext_vector_type(8)));

typedef unsigned int uint32x4_t __attribute__((ext_vector_type(4)));
typedef int          int32x4_t  __attribute__((ext_vector_type(4)));
typedef int          int32x8_t  __attribute__((ext_vector_type(8)));

#if __has_builtin(__builtin_amdgcn_tensor_load_to_lds) && \
    __has_builtin(__builtin_amdgcn_s_wait_tensorcnt)
#define USE_TDM 1
#else
#define USE_TDM 0
#endif

// round-to-nearest-even f32 -> bf16 (bit pattern as ushort)
static __device__ __forceinline__ unsigned short f2bf(float f) {
    unsigned u = __float_as_uint(f);
    unsigned r = u + 0x7FFFu + ((u >> 16) & 1u);
    return (unsigned short)(r >> 16);
}

// Generic pointer into LDS -> raw LDS byte offset (low 32 bits of the address)
static __device__ __forceinline__ unsigned lds_off(const void* p) {
    return (unsigned)(uintptr_t)p;
}

#if USE_TDM
// TDM: DMA a 2-D tile of `rows` x 32 bf16 elements (row stride = ldK elements)
// from global memory into LDS at byte offset ldsAddr. Tracked by TENSORcnt.
// 6-arg builtin form (clang-23 / therock-10.0 headers).
static __device__ __forceinline__ void tdm_load_rows_x32(const unsigned short* gsrc,
                                                         unsigned ldsAddr,
                                                         unsigned rows,
                                                         unsigned ldK) {
    unsigned long long ga = (unsigned long long)(uintptr_t)gsrc;
    // D# group 0: count=1 (valid), lds_addr, global_addr[56:0], type=2 ("image")
    uint32x4_t g0;
    g0[0] = 1u;
    g0[1] = ldsAddr;
    g0[2] = (unsigned)(ga & 0xFFFFFFFFu);
    g0[3] = (unsigned)((ga >> 32) & 0x01FFFFFFu) | (2u << 30);
    // D# group 1: data_size=1 (2B), tensor_dim0=32, tensor_dim1=rows,
    // tile_dim0=32, tile_dim1=rows, tensor_dim0_stride=ldK, no padding/iterate.
    const unsigned td0 = 32u, td1 = rows, t0 = 32u, t1 = rows;
    int32x8_t g1;
    g1[0] = (int)(1u << 16);                                   // data_size=2B
    g1[1] = (int)((td0 & 0xFFFFu) << 16);                      // tensor_dim0 lo
    g1[2] = (int)((td0 >> 16) | ((td1 & 0xFFFFu) << 16));      // dim0 hi | dim1 lo
    g1[3] = (int)((td1 >> 16) | (t0 << 16));                   // dim1 hi | tile_dim0
    g1[4] = (int)t1;                                           // tile_dim1 (tile_dim2=0)
    g1[5] = (int)ldK;                                          // dim0_stride[31:0]
    g1[6] = 0;                                                 // dim0_stride hi | dim1_stride lo
    g1[7] = 0;
    int32x4_t gz4 = {0, 0, 0, 0};
    int32x8_t gz8 = {0, 0, 0, 0, 0, 0, 0, 0};
    __builtin_amdgcn_tensor_load_to_lds(g0, g1, gz4, gz4, gz8, 0);
}
#endif

// WMMA 16-bit A/B fragment: 16 bf16 per lane (ISA 7.12.2 16-bit 16x32 layout).
// Lanes 0-15: row = lane, kb=0; lanes 16-31: row = lane-16, kb=8.
// elems 0..7 = K [kb,kb+8), elems 8..15 = K [kb+16,kb+24).
union Frag {
    v16bf v;
    unsigned short u[16];
    uint4 q2[2];
};

static __device__ __forceinline__ void load_frag(Frag& f, const unsigned short* p) {
    f.q2[0] = *(const uint4*)(p);
    f.q2[1] = *(const uint4*)(p + 16);
}

static __device__ __forceinline__ v8f wmma_bf16(const Frag& a, const Frag& b, v8f c) {
    return __builtin_amdgcn_wmma_f32_16x16x32_bf16(false, a.v, false, b.v,
                                                   (short)0, c, false, false);
}

// ---------------------------------------------------------------------------
// f32 -> bf16 conversion
// ---------------------------------------------------------------------------
__global__ __launch_bounds__(256) void cvt_f32_bf16(const float* __restrict__ in,
                                                    unsigned short* __restrict__ out,
                                                    int n) {
    int i = blockIdx.x * 256 + threadIdx.x;
    if (i < n) out[i] = f2bf(in[i]);
}

// ---------------------------------------------------------------------------
// GEMM: out = A(MxK,bf16) @ W(NxK,bf16)^T + bias
// MODE 0: bf16 out, head-split (B,H,S,DH); MODE 1: head-split-T (B,H,DH,S);
// MODE 2: f32 out, plain (M,N).
// Block = 8 waves; block tile 128(m) x 64(n); wave tile 16 x 64.
// The shared 64x32 W tile is streamed into double-buffered LDS by the TDM
// (one descriptor per k-chunk issued by wave 0, s_wait_tensorcnt + barrier),
// while A fragments come straight from global per wave.
// ---------------------------------------------------------------------------
template <int MODE>
__global__ __launch_bounds__(256) void gemm_bf16_wmma(const unsigned short* __restrict__ A,
                                                      const unsigned short* __restrict__ W,
                                                      const float* __restrict__ bias,
                                                      void* __restrict__ outp,
                                                      int M, int N, int K) {
    __shared__ unsigned short sw[2][64 * 32];  // 8 KB double-buffered B tile

    const int lane = threadIdx.x & 31;
    const int wave = threadIdx.x >> 5;
    const int m0 = blockIdx.x * 128 + wave * 16;
    const int n0 = blockIdx.y * 64;
    const int halfsel = lane >> 4;
    const int lm = lane & 15;
    const int kboff = halfsel * 8;

    const unsigned short* arow  = A + (size_t)(m0 + lm) * K + kboff;
    const unsigned short* wtile = W + (size_t)n0 * K;

    v8f zero = {0.f, 0.f, 0.f, 0.f, 0.f, 0.f, 0.f, 0.f};
    v8f acc0 = zero, acc1 = zero, acc2 = zero, acc3 = zero;

    const int nchunk = K / 32;
#if USE_TDM
    if (wave == 0) tdm_load_rows_x32(wtile, lds_off(&sw[0][0]), 64u, (unsigned)K);
#endif
    for (int i = 0; i < nchunk; ++i) {
        const int k0 = i * 32;
#if USE_TDM
        if (wave == 0) {
            if (i + 1 < nchunk) {
                tdm_load_rows_x32(wtile + (size_t)(k0 + 32),
                                  lds_off(&sw[(i + 1) & 1][0]), 64u, (unsigned)K);
                __builtin_amdgcn_s_wait_tensorcnt((short)1);
            } else {
                __builtin_amdgcn_s_wait_tensorcnt((short)0);
            }
        }
        __syncthreads();
#else
        {   // cooperative copy fallback: 256 threads x 8 elems = 64x32 tile
            const unsigned short* src = wtile + k0;
            int row = threadIdx.x >> 2, col = (threadIdx.x & 3) * 8;
            *(uint4*)&sw[i & 1][row * 32 + col] = *(const uint4*)(src + (size_t)row * K + col);
            __syncthreads();
        }
#endif
        const unsigned short* sb = &sw[i & 1][0];
        Frag a, b0, b1, b2, b3;
        load_frag(a, arow + k0);
        load_frag(b0, sb + (0 * 16 + lm) * 32 + kboff);
        load_frag(b1, sb + (1 * 16 + lm) * 32 + kboff);
        load_frag(b2, sb + (2 * 16 + lm) * 32 + kboff);
        load_frag(b3, sb + (3 * 16 + lm) * 32 + kboff);
        __builtin_prefetch(arow + k0 + 32, 0, 0);
        acc0 = wmma_bf16(a, b0, acc0);
        acc1 = wmma_bf16(a, b1, acc1);
        acc2 = wmma_bf16(a, b2, acc2);
        acc3 = wmma_bf16(a, b3, acc3);
        __syncthreads();
    }

    v8f accs[4] = {acc0, acc1, acc2, acc3};
#pragma unroll
    for (int j = 0; j < 4; ++j) {
#pragma unroll
        for (int r = 0; r < 8; ++r) {
            int m = m0 + r + halfsel * 8;
            int n = n0 + j * 16 + lm;
            float val = accs[j][r] + bias[n];
            if (MODE == 0) {
                int b = m >> 11, s = m & (SS - 1);
                int h = n >> 6,  d = n & (DH - 1);
                ((unsigned short*)outp)[((size_t)(b * HH + h) * SS + s) * DH + d] = f2bf(val);
            } else if (MODE == 1) {
                int b = m >> 11, s = m & (SS - 1);
                int h = n >> 6,  d = n & (DH - 1);
                ((unsigned short*)outp)[((size_t)(b * HH + h) * DH + d) * SS + s] = f2bf(val);
            } else {
                ((float*)outp)[(size_t)m * N + n] = val;
            }
        }
    }
}

// ---------------------------------------------------------------------------
// scores = (Qh @ Kh^T) * 1/sqrt(DH), one (b,h). Qh,Kh: (S,DH) bf16. Sc: (S,S) f32.
// Wave tile: 16(q) x 64(k); K-loop is only DH/32 = 2 steps -> direct loads.
// ---------------------------------------------------------------------------
__global__ __launch_bounds__(256) void scores_wmma(const unsigned short* __restrict__ Qh,
                                                   const unsigned short* __restrict__ Kh,
                                                   float* __restrict__ Sc) {
    const int lane = threadIdx.x & 31;
    const int wave = threadIdx.x >> 5;
    const int q0 = blockIdx.x * 128 + wave * 16;
    const int k0 = blockIdx.y * 64;
    const int halfsel = lane >> 4;
    const int lm = lane & 15;
    const int kboff = halfsel * 8;

    const unsigned short* arow = Qh + (size_t)(q0 + lm) * DH + kboff;
    const unsigned short* brow = Kh + (size_t)(k0 + lm) * DH + kboff;

    v8f zero = {0.f, 0.f, 0.f, 0.f, 0.f, 0.f, 0.f, 0.f};
    v8f acc0 = zero, acc1 = zero, acc2 = zero, acc3 = zero;

#pragma unroll
    for (int kk = 0; kk < DH; kk += 32) {
        Frag a, b0, b1, b2, b3;
        load_frag(a, arow + kk);
        load_frag(b0, brow + kk);
        load_frag(b1, brow + 16 * DH + kk);
        load_frag(b2, brow + 32 * DH + kk);
        load_frag(b3, brow + 48 * DH + kk);
        acc0 = wmma_bf16(a, b0, acc0);
        acc1 = wmma_bf16(a, b1, acc1);
        acc2 = wmma_bf16(a, b2, acc2);
        acc3 = wmma_bf16(a, b3, acc3);
    }

    v8f accs[4] = {acc0, acc1, acc2, acc3};
#pragma unroll
    for (int j = 0; j < 4; ++j) {
#pragma unroll
        for (int r = 0; r < 8; ++r) {
            int q = q0 + r + halfsel * 8;
            int k = k0 + j * 16 + lm;
            Sc[(size_t)q * SS + k] = accs[j][r] * 0.125f;  // 1/sqrt(64)
        }
    }
}

// ---------------------------------------------------------------------------
// Softmax over the QUERY axis (axis=-2 in reference): per column k, online
// max / sum(exp) down q; stores max and reciprocal sum. Adjacent threads own
// adjacent columns -> coalesced row reads.
// ---------------------------------------------------------------------------
__global__ __launch_bounds__(256) void softmax_col_stats(const float* __restrict__ Sc,
                                                         float* __restrict__ cmax,
                                                         float* __restrict__ rsum) {
    int k = blockIdx.x * 256 + threadIdx.x;
    float m = -3.4e38f, z = 0.f;
    for (int q = 0; q < SS; ++q) {
        float v = Sc[(size_t)q * SS + k];
        if (v > m) { z = z * __expf(m - v) + 1.f; m = v; }
        else       { z += __expf(v - m); }
    }
    cmax[k] = m;
    rsum[k] = 1.f / z;
}

// ---------------------------------------------------------------------------
// out = softmax_colnorm(Sc) @ V for one (b,h). Vt: (DH,S) bf16.
// Wave tile 16(q) x 64(d); attn weights built on the fly (exp * rsum -> bf16).
// The 64x32 V tile (shared by all 8 waves) is TDM-streamed into LDS,
// double-buffered across the 64 k-chunks.
// ---------------------------------------------------------------------------
__global__ __launch_bounds__(256) void av_wmma(const float* __restrict__ Sc,
                                               const float* __restrict__ cmax,
                                               const float* __restrict__ rsum,
                                               const unsigned short* __restrict__ Vt,
                                               unsigned short* __restrict__ AO,
                                               int b, int h) {
    __shared__ unsigned short sv[2][64 * 32];

    const int lane = threadIdx.x & 31;
    const int wave = threadIdx.x >> 5;
    const int q0 = blockIdx.x * 128 + wave * 16;
    const int halfsel = lane >> 4;
    const int lm = lane & 15;
    const int kboff = halfsel * 8;
    const int q = q0 + lm;

    v8f zero = {0.f, 0.f, 0.f, 0.f, 0.f, 0.f, 0.f, 0.f};
    v8f acc0 = zero, acc1 = zero, acc2 = zero, acc3 = zero;

    const int nchunk = SS / 32;
#if USE_TDM
    if (wave == 0) tdm_load_rows_x32(Vt, lds_off(&sv[0][0]), 64u, (unsigned)SS);
#endif
    for (int i = 0; i < nchunk; ++i) {
        const int k0 = i * 32;
#if USE_TDM
        if (wave == 0) {
            if (i + 1 < nchunk) {
                tdm_load_rows_x32(Vt + (size_t)(k0 + 32),
                                  lds_off(&sv[(i + 1) & 1][0]), 64u, (unsigned)SS);
                __builtin_amdgcn_s_wait_tensorcnt((short)1);
            } else {
                __builtin_amdgcn_s_wait_tensorcnt((short)0);
            }
        }
        __syncthreads();
#else
        {
            const unsigned short* src = Vt + k0;
            int row = threadIdx.x >> 2, col = (threadIdx.x & 3) * 8;
            *(uint4*)&sv[i & 1][row * 32 + col] = *(const uint4*)(src + (size_t)row * SS + col);
            __syncthreads();
        }
#endif
        const int kb = k0 + kboff;
        const float* srow = Sc + (size_t)q * SS + kb;

        Frag a;
#pragma unroll
        for (int e = 0; e < 8; ++e) {
            float p0 = __expf(srow[e]      - cmax[kb + e])      * rsum[kb + e];
            float p1 = __expf(srow[e + 16] - cmax[kb + 16 + e]) * rsum[kb + 16 + e];
            a.u[e]     = f2bf(p0);
            a.u[e + 8] = f2bf(p1);
        }

        const unsigned short* sb = &sv[i & 1][0];
        Frag b0, b1, b2, b3;
        load_frag(b0, sb + (0 * 16 + lm) * 32 + kboff);
        load_frag(b1, sb + (1 * 16 + lm) * 32 + kboff);
        load_frag(b2, sb + (2 * 16 + lm) * 32 + kboff);
        load_frag(b3, sb + (3 * 16 + lm) * 32 + kboff);
        __builtin_prefetch(srow + 32, 0, 0);

        acc0 = wmma_bf16(a, b0, acc0);
        acc1 = wmma_bf16(a, b1, acc1);
        acc2 = wmma_bf16(a, b2, acc2);
        acc3 = wmma_bf16(a, b3, acc3);
        __syncthreads();
    }

    v8f accs[4] = {acc0, acc1, acc2, acc3};
#pragma unroll
    for (int j = 0; j < 4; ++j) {
#pragma unroll
        for (int r = 0; r < 8; ++r) {
            int qq = q0 + r + halfsel * 8;
            int d = j * 16 + lm;
            AO[(size_t)(b * SS + qq) * DD + h * DH + d] = f2bf(accs[j][r]);
        }
    }
}

// ---------------------------------------------------------------------------
// Host-side orchestration
// ---------------------------------------------------------------------------
extern "C" void kernel_launch(void* const* d_in, const int* in_sizes, int n_in,
                              void* d_out, int out_size, void* d_ws, size_t ws_size,
                              hipStream_t stream) {
    (void)in_sizes; (void)n_in; (void)out_size; (void)ws_size;

    const float* query = (const float*)d_in[0];
    const float* keyin = (const float*)d_in[1];
    const float* value = (const float*)d_in[2];
    const float* Wq = (const float*)d_in[3];
    const float* bq = (const float*)d_in[4];
    const float* Wk = (const float*)d_in[5];
    const float* bk = (const float*)d_in[6];
    const float* Wv = (const float*)d_in[7];
    const float* bv = (const float*)d_in[8];
    const float* Wo = (const float*)d_in[9];
    const float* bo = (const float*)d_in[10];
    float* out = (float*)d_out;

    // Workspace layout (bf16 unless noted); total ~67 MB.
    const size_t eQKV = (size_t)BB * HH * SS * DH;
    const size_t eX   = (size_t)BB * SS * DD;
    const size_t eW   = (size_t)DD * DD;
    unsigned short* Qb  = (unsigned short*)d_ws;
    unsigned short* Kb  = Qb + eQKV;
    unsigned short* Vtb = Kb + eQKV;
    unsigned short* Xb  = Vtb + eQKV;
    unsigned short* Wqb = Xb + eX;
    unsigned short* Wkb = Wqb + eW;
    unsigned short* Wvb = Wkb + eW;
    unsigned short* Wob = Wvb + eW;
    unsigned short* AOb = Wob + eW;
    float* Sc   = (float*)(AOb + eX);   // (S,S) f32, reused per (b,h)
    float* cmax = Sc + (size_t)SS * SS;
    float* rsum = cmax + SS;

    const int M = BB * SS, N = DD, K = DD;
    dim3 blk(256);
    dim3 gemmGrid(M / 128, N / 64);

    cvt_f32_bf16<<<(int)((eW + 255) / 256), blk, 0, stream>>>(Wq, Wqb, (int)eW);
    cvt_f32_bf16<<<(int)((eW + 255) / 256), blk, 0, stream>>>(Wk, Wkb, (int)eW);
    cvt_f32_bf16<<<(int)((eW + 255) / 256), blk, 0, stream>>>(Wv, Wvb, (int)eW);
    cvt_f32_bf16<<<(int)((eW + 255) / 256), blk, 0, stream>>>(Wo, Wob, (int)eW);

    cvt_f32_bf16<<<(int)((eX + 255) / 256), blk, 0, stream>>>(query, Xb, (int)eX);
    gemm_bf16_wmma<0><<<gemmGrid, blk, 0, stream>>>(Xb, Wqb, bq, Qb, M, N, K);
    cvt_f32_bf16<<<(int)((eX + 255) / 256), blk, 0, stream>>>(keyin, Xb, (int)eX);
    gemm_bf16_wmma<0><<<gemmGrid, blk, 0, stream>>>(Xb, Wkb, bk, Kb, M, N, K);
    cvt_f32_bf16<<<(int)((eX + 255) / 256), blk, 0, stream>>>(value, Xb, (int)eX);
    gemm_bf16_wmma<1><<<gemmGrid, blk, 0, stream>>>(Xb, Wvb, bv, Vtb, M, N, K);

    dim3 scGrid(SS / 128, SS / 64);
    dim3 avGrid(SS / 128);
    for (int bh = 0; bh < BB * HH; ++bh) {
        const unsigned short* Qh = Qb + (size_t)bh * SS * DH;
        const unsigned short* Kh = Kb + (size_t)bh * SS * DH;
        const unsigned short* Vh = Vtb + (size_t)bh * DH * SS;
        scores_wmma<<<scGrid, blk, 0, stream>>>(Qh, Kh, Sc);
        softmax_col_stats<<<SS / 256, blk, 0, stream>>>(Sc, cmax, rsum);
        av_wmma<<<avGrid, blk, 0, stream>>>(Sc, cmax, rsum, Vh, AOb, bh / HH, bh % HH);
    }

    gemm_bf16_wmma<2><<<gemmGrid, blk, 0, stream>>>(AOb, Wob, bo, out, M, N, K);
}
